// UnifiedRetriever_32710470926375
// MI455X (gfx1250) — matrix operand order repaired and checked by reading, not compiled, not run on hardware
//
#include <hip/hip_runtime.h>

// Problem constants (from reference setup_inputs)
#define BQ 16
#define CC 64
#define TX 128
#define TY 128
#define DD 768
#define KC 32            // K per WMMA chunk (bf16 16x16x32)
#define NKC (DD / KC)    // 24 chunks
#define LDSW 20          // padded row stride in dwords (16B aligned, conflict-friendly)

typedef __attribute__((ext_vector_type(8)))  float        v8f;
typedef __attribute__((ext_vector_type(16))) __bf16       v16bf;
typedef __attribute__((ext_vector_type(8)))  unsigned int v8u;
typedef __attribute__((ext_vector_type(4)))  float        f4;
typedef __attribute__((ext_vector_type(4)))  unsigned int u4;

// Round-to-nearest-even fp32 -> bf16 (as 16-bit pattern in low bits)
__device__ __forceinline__ unsigned bf16_rne(float f) {
  unsigned u = __builtin_bit_cast(unsigned, f);
  return (u + 0x7FFFu + ((u >> 16) & 1u)) >> 16;
}

// Split two fp32 into packed bf16 hi pair and bf16 lo (residual) pair.
__device__ __forceinline__ void split_pack(float a, float b, unsigned& hp, unsigned& lp) {
  unsigned ha = bf16_rne(a);
  unsigned hb = bf16_rne(b);
  float ra = a - __builtin_bit_cast(float, ha << 16);
  float rb = b - __builtin_bit_cast(float, hb << 16);
  hp = ha | (hb << 16);
  lp = bf16_rne(ra) | (bf16_rne(rb) << 16);
}

__global__ __launch_bounds__(256)
void hardattn_score_kernel(const float* __restrict__ xs,
                           const float* __restrict__ ys,
                           float* __restrict__ out) {
  __shared__ unsigned sBh[TY][LDSW];   // packed bf16-hi pairs: [row][k/2]
  __shared__ unsigned sBl[TY][LDSW];   // packed bf16-lo pairs
  __shared__ float    sRed[8];

  const int bc   = blockIdx.x;                 // b*CC + c
  const int tid  = threadIdx.x;
  const int wave = tid >> 5;
  const int lane = tid & 31;
  const int hx   = lane >> 4;                  // lane half (0/1)
  const int r    = lane & 15;

  const float* __restrict__ xsb  = xs + (size_t)(bc / CC) * TX * DD;
  const float* __restrict__ ysbc = ys + (size_t)bc * TY * DD;

  // A fragment source: this lane's row, offset by half*16 K values
  const float* aRow = xsb + (size_t)(wave * 16 + r) * DD + hx * 16;
  // Cooperative B staging: thread t handles 16 consecutive K of row t>>1
  const int   bRowIdx = tid >> 1;
  const int   bColBlk = tid & 1;                       // 0 or 1 (16 K values each)
  const float* bRow   = ysbc + (size_t)bRowIdx * DD + bColBlk * 16;

  v8f acc[8];
#pragma unroll
  for (int nt = 0; nt < 8; ++nt) acc[nt] = (v8f){0.f,0.f,0.f,0.f,0.f,0.f,0.f,0.f};

  for (int kc = 0; kc < NKC; ++kc) {
    // ---- cooperative fp32->bf16(hi,lo) conversion of ys chunk into LDS ----
    {
      const float* p = bRow + kc * KC;
      if (kc + 1 < NKC) __builtin_prefetch(p + KC, 0, 0);
      f4 f0 = *(const f4*)(p + 0);
      f4 f1 = *(const f4*)(p + 4);
      f4 f2 = *(const f4*)(p + 8);
      f4 f3 = *(const f4*)(p + 12);
      unsigned h[8], l[8];
      split_pack(f0.x, f0.y, h[0], l[0]);
      split_pack(f0.z, f0.w, h[1], l[1]);
      split_pack(f1.x, f1.y, h[2], l[2]);
      split_pack(f1.z, f1.w, h[3], l[3]);
      split_pack(f2.x, f2.y, h[4], l[4]);
      split_pack(f2.z, f2.w, h[5], l[5]);
      split_pack(f3.x, f3.y, h[6], l[6]);
      split_pack(f3.z, f3.w, h[7], l[7]);
      unsigned* dh = &sBh[bRowIdx][bColBlk * 8];
      unsigned* dl = &sBl[bRowIdx][bColBlk * 8];
      *(u4*)(dh + 0) = (u4){h[0], h[1], h[2], h[3]};
      *(u4*)(dh + 4) = (u4){h[4], h[5], h[6], h[7]};
      *(u4*)(dl + 0) = (u4){l[0], l[1], l[2], l[3]};
      *(u4*)(dl + 4) = (u4){l[4], l[5], l[6], l[7]};
    }
    __syncthreads();

    // ---- A fragment (16x32 bf16): per-lane split from global ----
    const float* pa = aRow + kc * KC;
    f4 a0 = *(const f4*)(pa + 0);
    f4 a1 = *(const f4*)(pa + 4);
    f4 a2 = *(const f4*)(pa + 8);
    f4 a3 = *(const f4*)(pa + 12);
    unsigned hA[8], lA[8];
    split_pack(a0.x, a0.y, hA[0], lA[0]);
    split_pack(a0.z, a0.w, hA[1], lA[1]);
    split_pack(a1.x, a1.y, hA[2], lA[2]);
    split_pack(a1.z, a1.w, hA[3], lA[3]);
    split_pack(a2.x, a2.y, hA[4], lA[4]);
    split_pack(a2.z, a2.w, hA[5], lA[5]);
    split_pack(a3.x, a3.y, hA[6], lA[6]);
    split_pack(a3.z, a3.w, hA[7], lA[7]);
    v8u ah, al;
#pragma unroll
    for (int i = 0; i < 8; ++i) { ah[i] = hA[i]; al[i] = lA[i]; }
    v16bf Ahi = __builtin_bit_cast(v16bf, ah);
    v16bf Alo = __builtin_bit_cast(v16bf, al);

    // ---- 8 N-tiles: B fragments from LDS (32x16 bf16: lane half holds 16 K) ----
#pragma unroll
    for (int nt = 0; nt < 8; ++nt) {
      const int row = nt * 16 + r;
      const unsigned* bhp = &sBh[row][hx * 8];
      const unsigned* blp = &sBl[row][hx * 8];
      u4 h0 = *(const u4*)(bhp + 0);
      u4 h1 = *(const u4*)(bhp + 4);
      u4 l0 = *(const u4*)(blp + 0);
      u4 l1 = *(const u4*)(blp + 4);
      v8u bh = __builtin_shufflevector(h0, h1, 0, 1, 2, 3, 4, 5, 6, 7);
      v8u bl = __builtin_shufflevector(l0, l1, 0, 1, 2, 3, 4, 5, 6, 7);
      v16bf Bhi = __builtin_bit_cast(v16bf, bh);
      v16bf Blo = __builtin_bit_cast(v16bf, bl);
      acc[nt] = __builtin_amdgcn_wmma_f32_16x16x32_bf16(false, Ahi, false, Bhi,
                                                        (short)0, acc[nt], false, false);
      acc[nt] = __builtin_amdgcn_wmma_f32_16x16x32_bf16(false, Ahi, false, Blo,
                                                        (short)0, acc[nt], false, false);
      acc[nt] = __builtin_amdgcn_wmma_f32_16x16x32_bf16(false, Alo, false, Bhi,
                                                        (short)0, acc[nt], false, false);
    }
    __syncthreads();
  }

  // ---- reduction: max over Ty (cols), sum over Tx (rows) ----
  v8f mx = acc[0];
#pragma unroll
  for (int nt = 1; nt < 8; ++nt) {
#pragma unroll
    for (int j = 0; j < 8; ++j) mx[j] = fmaxf(mx[j], acc[nt][j]);
  }

  float rowsum = 0.f;
#pragma unroll
  for (int j = 0; j < 8; ++j) {
    float v = mx[j];
    v = fmaxf(v, __shfl_xor(v, 1, 32));
    v = fmaxf(v, __shfl_xor(v, 2, 32));
    v = fmaxf(v, __shfl_xor(v, 4, 32));
    v = fmaxf(v, __shfl_xor(v, 8, 32));   // lane-halves now hold rowmax of their row
    rowsum += v;                           // sum 8 rows held by this lane-half
  }
  rowsum += __shfl_xor(rowsum, 16, 32);    // merge the two 8-row halves (16 rows/wave)

  if (lane == 0) sRed[wave] = rowsum;
  __syncthreads();
  if (tid == 0) {
    float s = 0.f;
#pragma unroll
    for (int w = 0; w < 8; ++w) s += sRed[w];
    out[bc] = s;
  }
}

extern "C" void kernel_launch(void* const* d_in, const int* in_sizes, int n_in,
                              void* d_out, int out_size, void* d_ws, size_t ws_size,
                              hipStream_t stream) {
  (void)in_sizes; (void)n_in; (void)out_size; (void)d_ws; (void)ws_size;
  const float* xs = (const float*)d_in[0];
  const float* ys = (const float*)d_in[1];
  float* out = (float*)d_out;
  dim3 grid(BQ * CC);
  dim3 block(256);
  hipLaunchKernelGGL(hardattn_score_kernel, grid, block, 0, stream, xs, ys, out);
}
